// EvolutionaryFeatureExtractor_69836168233624
// MI455X (gfx1250) — 8-bit matrix operand order repaired and checked by reading, compiled, not run on hardware
//
#include <hip/hip_runtime.h>
#include <hip/hip_bf16.h>
#include <math.h>

#define N_SEQS   2048
#define SEQ_LEN  512
#define N_AA     20
#define M_POS    100
#define NC       2048
#define KDIM     2048

typedef __attribute__((ext_vector_type(8))) int v8i;

__global__ void build_oht_kernel(const int* __restrict__ msa,
                                 unsigned int* __restrict__ OHT) {
    int t = blockIdx.x * blockDim.x + threadIdx.x;
    const int ND = KDIM / 4;
    if (t >= 103 * ND) return;
    int i  = t / ND;
    int nd = t % ND;
    int n0 = nd * 4;
    if (i < M_POS) {
        int m0 = msa[(size_t)(n0 + 0) * SEQ_LEN + i];
        int m1 = msa[(size_t)(n0 + 1) * SEQ_LEN + i];
        int m2 = msa[(size_t)(n0 + 2) * SEQ_LEN + i];
        int m3 = msa[(size_t)(n0 + 3) * SEQ_LEN + i];
#pragma unroll
        for (int a = 0; a < N_AA; ++a) {
            unsigned int d = (m0 == a ? 1u : 0u)
                           | (m1 == a ? 1u : 0u) << 8
                           | (m2 == a ? 1u : 0u) << 16
                           | (m3 == a ? 1u : 0u) << 24;
            OHT[(size_t)(i * N_AA + a) * ND + nd] = d;
        }
    } else {
#pragma unroll
        for (int a = 0; a < N_AA; ++a) {
            int c = i * N_AA + a;
            if (c < NC) OHT[(size_t)c * ND + nd] = 0u;
        }
    }
}

__global__ void pssm_kernel(const int* __restrict__ msa,
                            const float* __restrict__ pc,
                            float* __restrict__ pssm,
                            float* __restrict__ cons) {
    __shared__ int cnt[N_AA];
    int i   = blockIdx.x;
    int tid = threadIdx.x;
    if (tid < N_AA) cnt[tid] = 0;
    __syncthreads();
    for (int n = tid; n < N_SEQS; n += blockDim.x) {
        int v = msa[(size_t)n * SEQ_LEN + i];
        if (v < N_AA) atomicAdd(&cnt[v], 1);
    }
    __syncthreads();
    float pcnt = 0.01f * pc[0];
    if (tid < N_AA) {
        float freq = ((float)cnt[tid] + pcnt) / ((float)N_SEQS + pcnt * (float)N_AA);
        pssm[(size_t)i * N_AA + tid] = logf(freq * (float)N_AA + 1e-10f);
    }
    if (tid == 0) {
        float total = 0.f;
        for (int a = 0; a < N_AA; ++a) total += (float)cnt[a];
        float ts  = fmaxf(total, 1.0f);
        float ent = 0.f;
        for (int a = 0; a < N_AA; ++a) {
            float f = (float)cnt[a] / ts;
            ent -= f * log2f(f + 1e-10f);
        }
        cons[i] = (total > 0.f) ? (1.0f - ent / log2f((float)N_AA)) : 0.f;
    }
}

__global__ void zero_mi_kernel(float* __restrict__ p, int n) {
    int t = blockIdx.x * blockDim.x + threadIdx.x;
    if (t < n) p[t] = 0.f;
}

__global__ void gemm_iu8_kernel(const unsigned char* __restrict__ OHT,
                                int* __restrict__ C) {
    int gtid = blockIdx.x * blockDim.x + threadIdx.x;
    int wave = gtid >> 5;
    int lane = threadIdx.x & 31;
    int wr = wave >> 6;
    int wc = wave & 63;
    int row0 = wr * 32;
    int col0 = wc * 32;
    int half = lane >> 4;
    int l15  = lane & 15;

    v8i acc00 = {}, acc01 = {}, acc10 = {}, acc11 = {};

    const unsigned char* Abase0 = OHT + (size_t)(row0 + l15) * KDIM;
    const unsigned char* Abase1 = Abase0 + (size_t)16 * KDIM;
    const unsigned char* Bbase0 = OHT + (size_t)(col0 + l15) * KDIM;
    const unsigned char* Bbase1 = Bbase0 + (size_t)16 * KDIM;

    for (int k0 = 0; k0 < KDIM; k0 += 64) {
        v8i a0, a1, b0, b1;
#pragma unroll
        for (int v = 0; v < 8; ++v) {
            int ka = ((v & 4) << 3) + ((v & 2) << 3) + half * 8 + ((v & 1) << 2);
            int kb = ((v & 4) << 3) + ((v & 2) << 3) + half * 8 + ((v & 1) << 2);
            a0[v] = *(const int*)(Abase0 + k0 + ka);
            a1[v] = *(const int*)(Abase1 + k0 + ka);
            b0[v] = *(const int*)(Bbase0 + k0 + kb);
            b1[v] = *(const int*)(Bbase1 + k0 + kb);
        }
        acc00 = __builtin_amdgcn_wmma_i32_16x16x64_iu8(false, a0, false, b0, acc00, false, false);
        acc01 = __builtin_amdgcn_wmma_i32_16x16x64_iu8(false, a0, false, b1, acc01, false, false);
        acc10 = __builtin_amdgcn_wmma_i32_16x16x64_iu8(false, a1, false, b0, acc10, false, false);
        acc11 = __builtin_amdgcn_wmma_i32_16x16x64_iu8(false, a1, false, b1, acc11, false, false);
    }

#pragma unroll
    for (int d = 0; d < 8; ++d) {
        int r = d + half * 8;
        C[(size_t)(row0 + r)      * NC + col0 + l15]      = acc00[d];
        C[(size_t)(row0 + r)      * NC + col0 + 16 + l15] = acc01[d];
        C[(size_t)(row0 + 16 + r) * NC + col0 + l15]      = acc10[d];
        C[(size_t)(row0 + 16 + r) * NC + col0 + 16 + l15] = acc11[d];
    }
}

__global__ void mi_kernel(const int* __restrict__ C,
                          float* __restrict__ mi_out) {
    __shared__ float pi_s[4][N_AA];
    __shared__ float pj_s[4][N_AA];
    __shared__ float tot_s[4];
    int w    = threadIdx.x >> 5;
    int lane = threadIdx.x & 31;
    int pair = blockIdx.x * 4 + w;
    int i = pair / M_POS;
    int j = pair % M_POS;
    const int* base = C + (size_t)(i * N_AA) * NC + j * N_AA;

    if (lane < N_AA) {
        float pi = 0.f, pj = 0.f;
#pragma unroll
        for (int a = 0; a < N_AA; ++a) {
            pj += (float)base[(size_t)a * NC + lane];
            pi += (float)base[(size_t)lane * NC + a];
        }
        pi_s[w][lane] = pi;
        pj_s[w][lane] = pj;
    }
    __syncthreads();
    if (lane == 0) {
        float t = 0.f;
        for (int a = 0; a < N_AA; ++a) t += pj_s[w][a];
        tot_s[w] = t;
    }
    __syncthreads();
    float tot      = tot_s[w];
    float inv_tot  = 1.0f / fmaxf(tot, 1.0f);

    float partial = 0.f;
    if (lane < N_AA) {
        float pjb = pj_s[w][lane] * inv_tot;
#pragma unroll
        for (int a = 0; a < N_AA; ++a) {
            int v = base[(size_t)a * NC + lane];
            if (v > 0) {
                float p     = (float)v * inv_tot;
                float pia   = pi_s[w][a] * inv_tot;
                float denom = fmaxf(pia * pjb, 1e-30f);
                partial += p * log2f(p / denom + 1e-10f);
            }
        }
    }
#pragma unroll
    for (int off = 16; off > 0; off >>= 1)
        partial += __shfl_down(partial, off, 32);

    if (lane == 0) {
        float mi = (tot > 0.f && i != j) ? partial : 0.f;
        mi_out[(size_t)i * SEQ_LEN + j] = mi;
    }
}

extern "C" void kernel_launch(void* const* d_in, const int* in_sizes, int n_in,
                              void* d_out, int out_size, void* d_ws, size_t ws_size,
                              hipStream_t stream) {
    const int*   msa = (const int*)d_in[0];
    const float* pc  = (const float*)d_in[1];

    float* out  = (float*)d_out;
    float* pssm = out;
    float* cons = out + (size_t)SEQ_LEN * N_AA;
    float* mi   = cons + SEQ_LEN;

    unsigned char* OHT = (unsigned char*)d_ws;
    int*           C   = (int*)((char*)d_ws + (size_t)NC * KDIM);

    {
        int total = 103 * (KDIM / 4);
        build_oht_kernel<<<(total + 255) / 256, 256, 0, stream>>>(msa, (unsigned int*)OHT);
    }
    pssm_kernel<<<SEQ_LEN, 256, 0, stream>>>(msa, pc, pssm, cons);
    {
        int n = SEQ_LEN * SEQ_LEN;
        zero_mi_kernel<<<(n + 255) / 256, 256, 0, stream>>>(mi, n);
    }
    gemm_iu8_kernel<<<512, 256, 0, stream>>>(OHT, C);
    mi_kernel<<<(M_POS * M_POS) / 4, 128, 0, stream>>>(C, mi);
}
